// TgGCN_82660940579213
// MI455X (gfx1250) — matrix operand order repaired and checked by reading, not compiled, not run on hardware
//
#include <hip/hip_runtime.h>

typedef __attribute__((ext_vector_type(16))) _Float16 v16h;
typedef __attribute__((ext_vector_type(8)))  float    v8f;

static __device__ __forceinline__ uint32_t pack2h(float a, float b) {
  union { _Float16 h[2]; uint32_t u; } p;
  p.h[0] = (_Float16)a;
  p.h[1] = (_Float16)b;
  return p.u;
}

// ---------------------------------------------------------------- small kernels

__global__ void k_fill(float* __restrict__ p, float v, int n) {
  int i = blockIdx.x * blockDim.x + threadIdx.x;
  if (i < n) p[i] = v;
}

__global__ void k_zero4(float4* __restrict__ p, int n4) {
  int i = blockIdx.x * blockDim.x + threadIdx.x;
  if (i < n4) p[i] = make_float4(0.f, 0.f, 0.f, 0.f);
}

__global__ void k_deg_accum(const int* __restrict__ dst, float* __restrict__ deg, int E) {
  int i = blockIdx.x * blockDim.x + threadIdx.x;
  if (i < E) atomicAdd(&deg[dst[i]], 1.0f);
}

__global__ void k_norm(const float* __restrict__ deg, float* __restrict__ nrm,
                       float* __restrict__ dinv, int n) {
  int i = blockIdx.x * blockDim.x + threadIdx.x;
  if (i < n) {
    float d = deg[i];
    nrm[i]  = rsqrtf(d);
    dinv[i] = 1.0f / d;
  }
}

// ---------------------------------------------------------------- WMMA GEMM
// out[M x 128] = A[M x K] * W[K x 128] (+ bias), K in {128, 256}, M % 16 == 0.
// 8 waves per block; wave w computes rows [16*(8*blk+w) .. +16) x all 128 cols.
// W is repacked once per block into LDS as f16 pairs in this kernel's
// contiguous-k B-fragment order for V_WMMA_F32_16X16X32_F16 (same k order as A):
//   lane L -> N = L%16 ; lane group g = L/16 covers K-halves g*16..g*16+15,
//   VGPR j holds K = g*16 + 2j, 2j+1 (packed halves).
// A fragment (16-bit A 16x32): lane L -> M = L%16, group g = L/16;
//   pair j -> K = kb + g*16 + 2j (contiguous, same as B).
__global__ __launch_bounds__(256) void k_wmma_gemm_n128(
    const float* __restrict__ A, const float* __restrict__ W,
    const float* __restrict__ bias, float* __restrict__ out,
    int M, int K)
{
  __shared__ uint32_t bpack[8 * 8 * 32 * 8];  // [s][t][lane][pair], 64 KB (K=256)

  const int tid    = threadIdx.x;
  const int ksteps = K >> 5;                  // K/32

  // Cooperative repack of W[K x 128] -> f16 B fragments in LDS.
  const int total = ksteps * 2048;            // ksteps*8*32*8
  for (int idx = tid; idx < total; idx += 256) {
    const int j = idx & 7;
    const int L = (idx >> 3) & 31;
    const int t = (idx >> 8) & 7;
    const int s = idx >> 11;
    const int g = L >> 4;
    const int n = t * 16 + (L & 15);
    const int k = s * 32 + g * 16 + 2 * j;
    bpack[idx] = pack2h(W[(size_t)k * 128 + n], W[(size_t)(k + 1) * 128 + n]);
  }
  __syncthreads();

  const int wave   = tid >> 5;                // wave32
  const int lane   = tid & 31;
  const int rTile  = blockIdx.x * 8 + wave;
  const int mTiles = M >> 4;
  if (rTile >= mTiles) return;

  const int g    = lane >> 4;
  const int mloc = lane & 15;
  const float* arow = A + (size_t)(rTile * 16 + mloc) * K;

  v8f acc[8] = {};
  const uint4* b4 = (const uint4*)bpack;

  for (int s = 0; s < ksteps; ++s) {
    const int kb = s << 5;
    v16h a;
#pragma unroll
    for (int j = 0; j < 8; ++j) {
      const int k0 = kb + (g << 4) + (j << 1);
      const float2 f = *(const float2*)(arow + k0);
      a[2 * j]     = (_Float16)f.x;
      a[2 * j + 1] = (_Float16)f.y;
    }
#pragma unroll
    for (int t = 0; t < 8; ++t) {
      union { uint4 q[2]; v16h h; } bb;
      const int base = ((s * 8 + t) * 32 + lane) * 2;   // uint4 index
      bb.q[0] = b4[base];
      bb.q[1] = b4[base + 1];
      acc[t] = __builtin_amdgcn_wmma_f32_16x16x32_f16(
          false, a, false, bb.h, (short)0, acc[t], false, false);
    }
  }

  // C/D layout: VGPR r -> M = r + 8*g (within tile), N = lane%16 (within tile).
#pragma unroll
  for (int t = 0; t < 8; ++t) {
    const int col = t * 16 + mloc;
    const float bv = bias ? bias[col] : 0.0f;
#pragma unroll
    for (int r = 0; r < 8; ++r) {
      const size_t row = (size_t)rTile * 16 + r + 8 * g;
      out[row * 128 + col] = acc[t][r] + bv;
    }
  }
}

// ---------------------------------------------------------------- edge scatter
// One wave per edge; lane covers 4 consecutive columns of the 128-wide row.
// agg[dst] += norm[src]*norm[dst] * z[src]   (f32 atomics; agg & z are L2-resident)
__global__ __launch_bounds__(256) void k_scatter(
    const float* __restrict__ z, const float* __restrict__ nrm,
    const int* __restrict__ src, const int* __restrict__ dst,
    float* __restrict__ agg, int E)
{
  const int gid  = blockIdx.x * blockDim.x + threadIdx.x;
  const int e    = gid >> 5;
  const int lane = gid & 31;
  if (e >= E) return;
  const int s = src[e];
  const int d = dst[e];
  const float c = nrm[s] * nrm[d];
  const float4 v = *(const float4*)(z + (size_t)s * 128 + lane * 4);
  float* p = agg + (size_t)d * 128 + lane * 4;
  atomicAdd(p + 0, c * v.x);
  atomicAdd(p + 1, c * v.y);
  atomicAdd(p + 2, c * v.z);
  atomicAdd(p + 3, c * v.w);
}

// ---------------------------------------------------------------- combine
// out = agg + z * deg_inv[row] + bias[col]  (optional ReLU), vectorized float4.
__global__ void k_combine(const float4* __restrict__ agg, const float4* __restrict__ z,
                          const float* __restrict__ dinv, const float4* __restrict__ bias,
                          float4* __restrict__ out, int n, int relu)
{
  const int i = blockIdx.x * blockDim.x + threadIdx.x;   // over n*32 float4s
  if (i >= n * 32) return;
  const int row = i >> 5;
  const int c4  = i & 31;
  const float  di = dinv[row];
  const float4 a  = agg[i];
  const float4 zz = z[i];
  const float4 b  = bias[c4];
  float4 o;
  o.x = a.x + zz.x * di + b.x;
  o.y = a.y + zz.y * di + b.y;
  o.z = a.z + zz.z * di + b.z;
  o.w = a.w + zz.w * di + b.w;
  if (relu) {
    o.x = fmaxf(o.x, 0.f); o.y = fmaxf(o.y, 0.f);
    o.z = fmaxf(o.z, 0.f); o.w = fmaxf(o.w, 0.f);
  }
  out[i] = o;
}

// ---------------------------------------------------------------- launch

extern "C" void kernel_launch(void* const* d_in, const int* in_sizes, int n_in,
                              void* d_out, int out_size, void* d_ws, size_t ws_size,
                              hipStream_t stream) {
  const float* x  = (const float*)d_in[0];
  const int*   ei = (const int*)d_in[1];
  const float* Wp = (const float*)d_in[2];
  const float* bp = (const float*)d_in[3];
  const float* W1 = (const float*)d_in[4];
  const float* b1 = (const float*)d_in[5];
  const float* W2 = (const float*)d_in[6];
  const float* b2 = (const float*)d_in[7];

  const int N = in_sizes[0] / 256;   // nodes (100000)
  const int E = in_sizes[1] / 2;     // edges (1600000)
  const int* src = ei;
  const int* dst = ei + E;

  // Workspace layout (256B-aligned slices).
  char* ws = (char*)d_ws;
  auto take = [&](size_t bytes) {
    char* p = ws;
    ws += (bytes + 255) & ~(size_t)255;
    return p;
  };
  float* deg  = (float*)take((size_t)N * 4);
  float* nrm  = (float*)take((size_t)N * 4);
  float* dinv = (float*)take((size_t)N * 4);
  float* bufA = (float*)take((size_t)N * 128 * 4);   // h, then h1
  float* bufB = (float*)take((size_t)N * 128 * 4);   // z1, then z2
  float* bufC = (float*)take((size_t)N * 128 * 4);   // agg (re-zeroed per use)

  const int T = 256;
  const int nBlk   = (N + T - 1) / T;
  const int eBlk   = (E + T - 1) / T;
  const int mTiles = N / 16;
  const int gBlk   = (mTiles + 7) / 8;
  const int n4     = N * 32;                         // float4 count of [N,128]
  const int c4Blk  = (n4 + T - 1) / T;
  const int sBlk   = (int)(((size_t)E * 32 + T - 1) / T);

  // degrees / norms
  k_fill<<<nBlk, T, 0, stream>>>(deg, 1.0f, N);
  k_deg_accum<<<eBlk, T, 0, stream>>>(dst, deg, E);
  k_norm<<<nBlk, T, 0, stream>>>(deg, nrm, dinv, N);

  // h = x @ Wp + bp
  k_wmma_gemm_n128<<<gBlk, T, 0, stream>>>(x, Wp, bp, bufA, N, 256);

  // layer 1: z1 = h @ W1 ; agg ; h1 = relu(agg + z1*dinv + b1)
  k_wmma_gemm_n128<<<gBlk, T, 0, stream>>>(bufA, W1, nullptr, bufB, N, 128);
  k_zero4<<<c4Blk, T, 0, stream>>>((float4*)bufC, n4);
  k_scatter<<<sBlk, T, 0, stream>>>(bufB, nrm, src, dst, bufC, E);
  k_combine<<<c4Blk, T, 0, stream>>>((const float4*)bufC, (const float4*)bufB,
                                     dinv, (const float4*)b1, (float4*)bufA, N, 1);

  // layer 2: z2 = h1 @ W2 ; agg ; out = agg + z2*dinv + b2
  k_wmma_gemm_n128<<<gBlk, T, 0, stream>>>(bufA, W2, nullptr, bufB, N, 128);
  k_zero4<<<c4Blk, T, 0, stream>>>((float4*)bufC, n4);
  k_scatter<<<sBlk, T, 0, stream>>>(bufB, nrm, src, dst, bufC, E);
  k_combine<<<c4Blk, T, 0, stream>>>((const float4*)bufC, (const float4*)bufB,
                                     dinv, (const float4*)b2, (float4*)d_out, N, 0);
}
